// Model_61229053772034
// MI455X (gfx1250) — matrix operand order repaired and checked by reading, not compiled, not run on hardware
//
#include <hip/hip_runtime.h>

// ---------- problem constants ----------
#define BATCH 16
#define SPD   24          // input spatial dim
#define OD    26          // output spatial dim (SPD + 2)
#define CI    64
#define CO    128
#define NTAPS 27          // 3*3*3
#define GROUPS 8
#define CPG   16          // channels per group (CO/GROUPS)
#define POS   (OD*OD*OD)  // 17576 spatial positions per batch
#define ROWW  35          // padded LDS row width: w index range [-2, 32] -> 35
#define NROWS 12          // 3 d-rows x 4 h-rows (two output h per workgroup)
#define WB_ELEMS (NTAPS*CO*CI)        // 221184 bf16 weights
#define WB_BYTES ((size_t)WB_ELEMS*2) // 442368
#define XBF_OFF_BYTES (WB_BYTES + 1024)   // bf16 x after weights + stats
#define XN_ELEMS (SPD*SPD*SPD*CI)         // 884736 elems per batch

typedef __attribute__((ext_vector_type(4)))  unsigned int u32x4;
typedef __attribute__((ext_vector_type(16))) __bf16       bf16x16;
typedef __attribute__((ext_vector_type(8)))  float        f32x8;

union FragBF { u32x4 u[2]; bf16x16 v; };

__device__ __forceinline__ unsigned int f2bf(float f) {
    unsigned int u = __builtin_bit_cast(unsigned int, f);
    unsigned int r = u + 0x7FFFu + ((u >> 16) & 1u);   // round-to-nearest-even
    return r >> 16;
}

// ---------------------------------------------------------------------------
// K0: x f32 -> bf16 (one pass; conv kernel then async-copies bf16 directly
// to LDS, and the 6x-overlapping row windows re-read half the bytes from L2).
// ---------------------------------------------------------------------------
__global__ __launch_bounds__(256) void xconv_kernel(
    const float* __restrict__ x, unsigned short* __restrict__ xbf)
{
    const size_t total4 = (size_t)BATCH * XN_ELEMS / 4;
    size_t idx = (size_t)blockIdx.x * 256 + threadIdx.x;
    if (idx >= total4) return;
    const float4 v = *(const float4*)(x + idx * 4);
    unsigned int lo = f2bf(v.x) | (f2bf(v.y) << 16);
    unsigned int hi = f2bf(v.z) | (f2bf(v.w) << 16);
    *(uint2*)(xbf + idx * 4) = make_uint2(lo, hi);
}

// ---------------------------------------------------------------------------
// K1: flip + transpose + f32->bf16 convert weights.
// dst: wb[tap][co][ci] bf16, tap=(a*3+b)*3+c, value = w[2-a][2-b][2-c][ci][co]
// ---------------------------------------------------------------------------
__global__ void wconv_kernel(const float* __restrict__ w, unsigned short* __restrict__ wb) {
    int idx = blockIdx.x * 256 + threadIdx.x;
    if (idx >= WB_ELEMS) return;
    int ci  = idx & (CI - 1);
    int t   = idx >> 6;
    int co  = t & (CO - 1);
    int tap = t >> 7;
    int a = tap / 9, b = (tap / 3) % 3, c = tap % 3;
    int src = ((((2 - a) * 3 + (2 - b)) * 3 + (2 - c)) * CI + ci) * CO + co;
    wb[idx] = (unsigned short)f2bf(w[src]);
}

// ---------------------------------------------------------------------------
// K2: implicit-GEMM transposed-conv + ReLU via v_wmma_f32_16x16x32_bf16.
// One workgroup per (n, do, h-pair): y[n,do,hb..hb+1, 0..25, 0..127].
// 8 waves; wave g owns co-slice [16g,16g+16) (== GN group g); 4 accumulators
// (2 h-rows x 2 w-tiles) share each B fragment -> 216 WMMAs/wave.
// A rows staged via CDNA5 GLOBAL_LOAD_ASYNC_TO_LDS_B128 (ASYNCcnt).
// ---------------------------------------------------------------------------
__global__ __launch_bounds__(256) void conv_wmma_kernel(
    const unsigned short* __restrict__ xbf, const unsigned short* __restrict__ wb,
    float* __restrict__ y)
{
    __shared__ unsigned short xrow[NROWS * ROWW * CI];   // 53760 B

    const int hb  = blockIdx.x * 2;   // output h-pair base
    const int dd  = blockIdx.y;
    const int n   = blockIdx.z;
    const int tid = threadIdx.x;

    // -- zero LDS (halo padding + out-of-image rows), 16B stores --
    {
        u32x4 zz = {};
        u32x4* z = (u32x4*)xrow;
        for (int i = tid; i < NROWS * ROWW * CI * 2 / 16; i += 256) z[i] = zz;
    }
    __syncthreads();   // zeros land before async writes to overlapping region

    // -- async-copy the 12 (a,hh) bf16 input rows: 192 x 16B chunks per row --
    {
        const unsigned short* sbase = xbf + (size_t)n * XN_ELEMS;  // uniform
        for (int i = tid; i < NROWS * 192; i += 256) {
            int r  = i / 192, ch = i - r * 192;
            int a  = r >> 2, hh = r & 3;
            int di = dd - 2 + a, hi = hb - 2 + hh;
            if (di >= 0 && di < SPD && hi >= 0 && hi < SPD) {
                // LDS byte offset: aperture lives in addr[63:32] -> low 32 bits
                unsigned int ldsb =
                    (unsigned int)(unsigned long long)&xrow[(r * ROWW + 2) * CI]
                    + (unsigned int)(ch * 16);
                int voff = ((di * SPD + hi) * SPD * CI) * 2 + ch * 16;
                asm volatile("global_load_async_to_lds_b128 %0, %1, %2"
                             :: "v"(ldsb), "v"(voff), "s"(sbase)
                             : "memory");
            }
        }
    }
    asm volatile("s_wait_asynccnt 0" ::: "memory");
    __syncthreads();

    const int lane    = tid & 31;
    const int wave    = tid >> 5;
    const int lane15  = lane & 15;
    const int khalf   = lane >> 4;
    const int co_lane = wave * 16 + lane15;

    f32x8 acc00 = {}, acc01 = {}, acc10 = {}, acc11 = {};
    const unsigned short* wbl = wb + (size_t)co_lane * CI + khalf * 8;

    for (int tap = 0; tap < NTAPS; ++tap) {
        const int a  = tap / 9;
        const int rb = tap / 3;          // a*3+b
        const int b  = rb - a * 3;
        const int c  = tap - rb * 3;
        const int r0 = a * 4 + b;        // LDS row for output h = hb
        const unsigned short* as0 = &xrow[(r0 * ROWW + lane15 + c) * CI + khalf * 8];
        const unsigned short* as1 = as0 + ROWW * CI;     // output h = hb+1
        const unsigned short* wt  = wbl + (size_t)tap * (CO * CI);
        #pragma unroll
        for (int kk = 0; kk < CI; kk += 32) {
            // B fragment: two 16B chunks at ci, ci+16 (lane<16: K=0..7,16..23)
            FragBF bfr;
            const u32x4* q = (const u32x4*)(wt + kk);
            bfr.u[0] = q[0]; bfr.u[1] = q[2];
            // A fragments: ISA 16-bit A layout -> two 16B chunks at ci, ci+16
            FragBF a00, a01, a10, a11;
            const u32x4* p;
            p = (const u32x4*)(as0 + kk);           a00.u[0] = p[0]; a00.u[1] = p[2];
            p = (const u32x4*)(as0 + 16 * CI + kk); a01.u[0] = p[0]; a01.u[1] = p[2];
            p = (const u32x4*)(as1 + kk);           a10.u[0] = p[0]; a10.u[1] = p[2];
            p = (const u32x4*)(as1 + 16 * CI + kk); a11.u[0] = p[0]; a11.u[1] = p[2];
            acc00 = __builtin_amdgcn_wmma_f32_16x16x32_bf16(
                false, a00.v, false, bfr.v, (short)0, acc00, false, false);
            acc01 = __builtin_amdgcn_wmma_f32_16x16x32_bf16(
                false, a01.v, false, bfr.v, (short)0, acc01, false, false);
            acc10 = __builtin_amdgcn_wmma_f32_16x16x32_bf16(
                false, a10.v, false, bfr.v, (short)0, acc10, false, false);
            acc11 = __builtin_amdgcn_wmma_f32_16x16x32_bf16(
                false, a11.v, false, bfr.v, (short)0, acc11, false, false);
        }
    }

    // -- epilogue: ReLU + store pre-norm y; D layout: VGPR v -> M = v + 8*khalf --
    #pragma unroll
    for (int h = 0; h < 2; ++h) {
        float* ybase = y + (((size_t)n * OD + dd) * OD + (hb + h)) * OD * CO + co_lane;
        f32x8& A0 = h ? acc10 : acc00;
        f32x8& A1 = h ? acc11 : acc01;
        #pragma unroll
        for (int v = 0; v < 8; ++v) {
            int wo0 = khalf * 8 + v;                       // 0..15: always valid
            ybase[(size_t)wo0 * CO] = fmaxf(A0[v], 0.0f);
            int wo1 = 16 + khalf * 8 + v;                  // 16..31: clip at 26
            if (wo1 < OD) ybase[(size_t)wo1 * CO] = fmaxf(A1[v], 0.0f);
        }
    }
}

// ---------------------------------------------------------------------------
// K3: deterministic per-(n,group) sum/sumsq -> mean, rstd.
// stats[0..127] = mean, stats[128..255] = rsqrt(var+eps)
// ---------------------------------------------------------------------------
__global__ __launch_bounds__(1024) void gn_stats_kernel(
    const float* __restrict__ y, float* __restrict__ stats)
{
    const int bg = blockIdx.x;           // n*8 + g
    const int n = bg >> 3, g = bg & 7;
    const float* base = y + (size_t)n * POS * CO + g * CPG;

    float s = 0.0f, ss = 0.0f;
    for (int i = threadIdx.x; i < POS * 4; i += 1024) {   // 4 float4 per position
        int pos = i >> 2, c4 = (i & 3) * 4;
        float4 v = *(const float4*)(base + (size_t)pos * CO + c4);
        s  += v.x + v.y + v.z + v.w;
        ss += v.x * v.x + v.y * v.y + v.z * v.z + v.w * v.w;
    }
    __shared__ float sh1[1024], sh2[1024];
    sh1[threadIdx.x] = s; sh2[threadIdx.x] = ss;
    __syncthreads();
    for (int o = 512; o > 0; o >>= 1) {
        if (threadIdx.x < o) {
            sh1[threadIdx.x] += sh1[threadIdx.x + o];
            sh2[threadIdx.x] += sh2[threadIdx.x + o];
        }
        __syncthreads();
    }
    if (threadIdx.x == 0) {
        const float cnt = (float)POS * (float)CPG;
        float m   = sh1[0] / cnt;
        float var = sh2[0] / cnt - m * m;
        stats[bg]       = m;
        stats[128 + bg] = rsqrtf(var + 1e-5f);
    }
}

// ---------------------------------------------------------------------------
// K4: in-place GroupNorm apply: out = (y - mean)*rstd*scale + bias
// ---------------------------------------------------------------------------
__global__ __launch_bounds__(256) void gn_apply_kernel(
    float* __restrict__ y, const float* __restrict__ stats,
    const float* __restrict__ scale, const float* __restrict__ bias)
{
    const size_t total4 = (size_t)BATCH * POS * CO / 4;
    size_t idx = (size_t)blockIdx.x * 256 + threadIdx.x;
    if (idx >= total4) return;
    size_t e = idx * 4;
    int co = (int)(e & (CO - 1));
    int n  = (int)(e / ((size_t)POS * CO));
    int bg = n * 8 + (co >> 4);
    float m = stats[bg], rs = stats[128 + bg];
    float4 v = *(float4*)(y + e);
    v.x = (v.x - m) * rs * scale[co + 0] + bias[co + 0];
    v.y = (v.y - m) * rs * scale[co + 1] + bias[co + 1];
    v.z = (v.z - m) * rs * scale[co + 2] + bias[co + 2];
    v.w = (v.w - m) * rs * scale[co + 3] + bias[co + 3];
    *(float4*)(y + e) = v;
}

// ---------------------------------------------------------------------------
extern "C" void kernel_launch(void* const* d_in, const int* in_sizes, int n_in,
                              void* d_out, int out_size, void* d_ws, size_t ws_size,
                              hipStream_t stream) {
    (void)in_sizes; (void)n_in; (void)out_size; (void)ws_size;
    const float* x      = (const float*)d_in[0];
    const float* w      = (const float*)d_in[1];
    const float* gscale = (const float*)d_in[2];
    const float* gbias  = (const float*)d_in[3];
    float* out = (float*)d_out;

    unsigned short* wb  = (unsigned short*)d_ws;
    float* stats        = (float*)((char*)d_ws + WB_BYTES);
    unsigned short* xbf = (unsigned short*)((char*)d_ws + XBF_OFF_BYTES);

    const size_t xtotal4 = (size_t)BATCH * XN_ELEMS / 4;
    xconv_kernel<<<(unsigned)((xtotal4 + 255) / 256), 256, 0, stream>>>(x, xbf);

    wconv_kernel<<<(WB_ELEMS + 255) / 256, 256, 0, stream>>>(w, wb);

    dim3 grid(OD / 2, OD, BATCH);
    conv_wmma_kernel<<<grid, 256, 0, stream>>>(xbf, wb, out);

    gn_stats_kernel<<<BATCH * GROUPS, 1024, 0, stream>>>(out, stats);

    const size_t total4 = (size_t)BATCH * POS * CO / 4;
    gn_apply_kernel<<<(unsigned)((total4 + 255) / 256), 256, 0, stream>>>(
        out, stats, gscale, gbias);
}
